// MambaUNet_34394098106760
// MI455X (gfx1250) — matrix operand order repaired and checked by reading, not compiled, not run on hardware
//
#include <hip/hip_runtime.h>
#include <hip/hip_bf16.h>

// =====================================================================
// Mamba-UNet forward for MI455X (gfx1250, wave32, WMMA).
// GEMM operands are pre-packed into the exact WMMA per-lane fragment
// layout so the inner loop is contiguous v16h (32B) loads + wmma.
// Each wave computes a 16x32 C tile (2 accumulators share one A frag).
// =====================================================================

typedef __attribute__((ext_vector_type(16))) _Float16 v16h;
typedef __attribute__((ext_vector_type(8)))  float    v8f;

#define LN_EPS 1e-5f

// ---------------------------------------------------------------------
// Fragment packing.
// A fragments: dst[((ti*nKc + kc)*32 + lane)*16 + e]
//   lane: row = ti*16 + (lane&15), half = lane>>4
//   e<8 : k = kc*32 + half*8 + e
//   e>=8: k = kc*32 + 16 + half*8 + (e-8)   (= kc*32 + 8 + half*8 + e)
// B fragments: dst[((tj*nKc + kc)*32 + lane)*16 + e]
//   n = tj*16 + (lane&15),  k same as A fragments (half = lane>>4)
// ---------------------------------------------------------------------
__global__ void k_packfrag_a(const float* __restrict__ src, long ld, long M, int K,
                             _Float16* __restrict__ dst, int nKc, long nTi) {
  long i = (long)blockIdx.x * blockDim.x + threadIdx.x;
  long total = nTi * (long)nKc * 512;
  if (i >= total) return;
  int  e    = (int)(i & 15);
  int  lane = (int)((i >> 4) & 31);
  long r    = i >> 9;
  int  kc   = (int)(r % nKc);
  long ti   = r / nKc;
  int  half = lane >> 4;
  long m = ti * 16 + (lane & 15);
  int  k = kc * 32 + ((e < 8) ? (half * 8 + e) : (8 + half * 8 + e));
  float v = (m < M && k < K) ? src[m * ld + k] : 0.f;
  dst[i] = (_Float16)v;
}

// trans=1: src is [N x K] (row stride ld); trans=0: src is [K x N] (row stride ld)
__global__ void k_packfrag_b(const float* __restrict__ src, long ld, int N, int K,
                             int trans, _Float16* __restrict__ dst, int nKc, int nTj) {
  long i = (long)blockIdx.x * blockDim.x + threadIdx.x;
  long total = (long)nTj * nKc * 512;
  if (i >= total) return;
  int  e    = (int)(i & 15);
  int  lane = (int)((i >> 4) & 31);
  long r    = i >> 9;
  int  kc   = (int)(r % nKc);
  int  tj   = (int)(r / nKc);
  int n = tj * 16 + (lane & 15);
  int k = kc * 32 + ((e < 8) ? ((lane >> 4) * 8 + e) : (8 + (lane >> 4) * 8 + e));
  float v = 0.f;
  if (n < N && k < K)
    v = trans ? src[(long)n * ld + k] : src[(long)k * ld + n];
  dst[i] = (_Float16)v;
}

// im2col for 3x3 pad-1 conv on HWC input, packed directly into A fragments.
// K = C*9, column k = ci*9 + kh*3 + kw (matches OIHW weight flattening).
__global__ void k_im2col_frag(const float* __restrict__ x, int H, int W, int C,
                              _Float16* __restrict__ dst, int nKc, long nTi) {
  long i = (long)blockIdx.x * blockDim.x + threadIdx.x;
  long total = nTi * (long)nKc * 512;
  if (i >= total) return;
  int  e    = (int)(i & 15);
  int  lane = (int)((i >> 4) & 31);
  long r    = i >> 9;
  int  kc   = (int)(r % nKc);
  long ti   = r / nKc;
  int  half = lane >> 4;
  long p = ti * 16 + (lane & 15);
  int  k = kc * 32 + ((e < 8) ? (half * 8 + e) : (8 + half * 8 + e));
  int  K = C * 9;
  float v = 0.f;
  if (p < (long)H * W && k < K) {
    int c = k / 9, rr = k % 9, kh = rr / 3, kw = rr % 3;
    int h = (int)(p / W) + kh - 1;
    int w = (int)(p % W) + kw - 1;
    if ((unsigned)h < (unsigned)H && (unsigned)w < (unsigned)W)
      v = x[((long)h * W + w) * C + c];
  }
  dst[i] = (_Float16)v;
}

// ---------------------------------------------------------------------
// WMMA GEMM on pre-packed fragments.
// C[M,N](f32,ldc) = A * B, one wave per 16x32 C tile.
// act: 0=none, 1=relu, 2=softplus
// ---------------------------------------------------------------------
__device__ __forceinline__ float ep_act(float v, int act) {
  if (act == 1) return fmaxf(v, 0.f);
  if (act == 2) return (v > 20.f) ? v : log1pf(__expf(v));
  return v;
}

__global__ void k_gemm_wmma(const _Float16* __restrict__ A,
                            const _Float16* __restrict__ B,
                            float* __restrict__ C,
                            long M, int N, int nKc, int nTj2, int ldc,
                            const float* __restrict__ bias, int act, long tiles) {
  long wid = (long)blockIdx.x * (blockDim.x >> 5) + (threadIdx.x >> 5);
  if (wid >= tiles) return;
  long ti  = wid / nTj2;
  int  tjw = (int)(wid % nTj2);
  int lane = threadIdx.x & 31;

  const v16h* Af = (const v16h*)A + ti * (long)nKc * 32 + lane;
  const v16h* B0 = (const v16h*)B + (long)(2 * tjw) * nKc * 32 + lane;
  const v16h* B1 = B0 + (long)nKc * 32;

  v8f acc0 = {}, acc1 = {};
  for (int kc = 0; kc < nKc; ++kc) {
    __builtin_prefetch((const void*)(Af + (long)(kc + 4) * 32), 0, 0);
    __builtin_prefetch((const void*)(B0 + (long)(kc + 4) * 32), 0, 0);
    __builtin_prefetch((const void*)(B1 + (long)(kc + 4) * 32), 0, 0);
    v16h a  = Af[(long)kc * 32];
    v16h b0 = B0[(long)kc * 32];
    v16h b1 = B1[(long)kc * 32];
    acc0 = __builtin_amdgcn_wmma_f32_16x16x32_f16(false, a, false, b0,
                                                  (short)0, acc0, false, false);
    acc1 = __builtin_amdgcn_wmma_f32_16x16x32_f16(false, a, false, b1,
                                                  (short)0, acc1, false, false);
  }

  int half = lane >> 4;
  int col  = lane & 15;
  int n0 = tjw * 32 + col;
  int n1 = n0 + 16;
  float bv0 = (bias && n0 < N) ? bias[n0] : 0.f;
  float bv1 = (bias && n1 < N) ? bias[n1] : 0.f;
#pragma unroll
  for (int r = 0; r < 8; ++r) {
    long m = ti * 16 + half * 8 + r;   // C layout: VGPR r -> M = r + half*8
    if (m < M) {
      if (n0 < N) C[m * (long)ldc + n0] = ep_act(acc0[r] + bv0, act);
      if (n1 < N) C[m * (long)ldc + n1] = ep_act(acc1[r] + bv1, act);
    }
  }
}

// ---------------------------------------------------------------------
// Elementwise / normalization / scan kernels
// ---------------------------------------------------------------------
__global__ void k_nchw_to_hwc(const float* __restrict__ x, float* __restrict__ y,
                              int C, long P) {
  long i = (long)blockIdx.x * blockDim.x + threadIdx.x;
  if (i >= (long)C * P) return;
  long c = i / P, p = i % P;
  y[p * C + c] = x[i];
}

__global__ void k_hwc_to_nchw(const float* __restrict__ x, float* __restrict__ y,
                              int C, long P) {
  long i = (long)blockIdx.x * blockDim.x + threadIdx.x;
  if (i >= (long)C * P) return;
  long c = i / P, p = i % P;
  y[i] = x[p * C + c];
}

// (H,W,C) -> (W,H,C)
__global__ void k_transpose_hwc(const float* __restrict__ x, float* __restrict__ y,
                                int H, int W, int C) {
  long i = (long)blockIdx.x * blockDim.x + threadIdx.x;
  if (i >= (long)H * W * C) return;
  int  c = (int)(i % C);
  long p = i / C;
  int h = (int)(p / W), w = (int)(p % W);
  y[((long)w * H + h) * C + c] = x[i];
}

__global__ void k_maxpool2(const float* __restrict__ x, float* __restrict__ y,
                           int H, int W, int C) {
  int H2 = H >> 1, W2 = W >> 1;
  long i = (long)blockIdx.x * blockDim.x + threadIdx.x;
  if (i >= (long)H2 * W2 * C) return;
  int  c = (int)(i % C);
  long p = i / C;
  int h = (int)(p / W2), w = (int)(p % W2);
  const float* b = x + ((long)(2 * h) * W + 2 * w) * C + c;
  float v = fmaxf(fmaxf(b[0], b[C]), fmaxf(b[(long)W * C], b[(long)W * C + C]));
  y[i] = v;
}

__global__ void k_add2(const float* a, const float* b, float* y, long n) {
  long i = (long)blockIdx.x * blockDim.x + threadIdx.x;
  if (i < n) y[i] = a[i] + b[i];
}

__global__ void k_add3(const float* a, const float* b, const float* c, float* y, long n) {
  long i = (long)blockIdx.x * blockDim.x + threadIdx.x;
  if (i < n) y[i] = a[i] + b[i] + c[i];
}

__global__ void k_concat(const float* __restrict__ a, const float* __restrict__ b,
                         float* __restrict__ y, long P, int Ca, int Cb) {
  int C = Ca + Cb;
  long i = (long)blockIdx.x * blockDim.x + threadIdx.x;
  if (i >= P * (long)C) return;
  long p = i / C;
  int  c = (int)(i % C);
  y[i] = (c < Ca) ? a[p * Ca + c] : b[p * Cb + (c - Ca)];
}

// Transpose-conv 2x2 stride-2 scatter: T[p, co*4+kh*2+kw] -> y[(2h+kh),(2w+kw),co]
__global__ void k_up_scatter(const float* __restrict__ T, const float* __restrict__ bias,
                             float* __restrict__ y, int H, int W, int Cout) {
  int c4 = Cout * 4;
  long i = (long)blockIdx.x * blockDim.x + threadIdx.x;
  if (i >= (long)H * W * c4) return;
  long p = i / c4;
  int  r = (int)(i % c4);
  int co = r >> 2, k = r & 3, kh = k >> 1, kw = k & 1;
  int h = (int)(p / W), w = (int)(p % W);
  int W2 = W * 2;
  y[((long)(2 * h + kh) * W2 + (2 * w + kw)) * Cout + co] = T[i] + bias[co];
}

__global__ void k_bn_stats(const float* __restrict__ x, long P, int C,
                           float* __restrict__ mean, float* __restrict__ var) {
  int c = blockIdx.x;
  float s = 0.f, s2 = 0.f;
  for (long p = threadIdx.x; p < P; p += blockDim.x) {
    float v = x[p * C + c];
    s += v; s2 += v * v;
  }
  __shared__ float sh[256], sh2[256];
  sh[threadIdx.x] = s; sh2[threadIdx.x] = s2;
  __syncthreads();
  for (int o = blockDim.x >> 1; o > 0; o >>= 1) {
    if ((int)threadIdx.x < o) { sh[threadIdx.x] += sh[threadIdx.x + o];
                                sh2[threadIdx.x] += sh2[threadIdx.x + o]; }
    __syncthreads();
  }
  if (threadIdx.x == 0) {
    float m = sh[0] / (float)P;
    mean[c] = m;
    var[c]  = sh2[0] / (float)P - m * m;
  }
}

__global__ void k_bn_apply(float* __restrict__ x, long P, int C,
                           const float* __restrict__ mean, const float* __restrict__ var,
                           const float* __restrict__ g, const float* __restrict__ b,
                           int relu) {
  long i = (long)blockIdx.x * blockDim.x + threadIdx.x;
  if (i >= P * (long)C) return;
  int c = (int)(i % C);
  float v = (x[i] - mean[c]) * rsqrtf(var[c] + LN_EPS) * g[c] + b[c];
  if (relu) v = fmaxf(v, 0.f);
  x[i] = v;
}

__global__ void k_layernorm(const float* __restrict__ x, const float* __restrict__ g,
                            const float* __restrict__ b, float* __restrict__ y, int C) {
  long t = blockIdx.x;
  const float* row = x + t * (long)C;
  float s = 0.f, s2 = 0.f;
  for (int c = threadIdx.x; c < C; c += blockDim.x) {
    float v = row[c]; s += v; s2 += v * v;
  }
  __shared__ float sh[256], sh2[256];
  sh[threadIdx.x] = s; sh2[threadIdx.x] = s2;
  __syncthreads();
  for (int o = blockDim.x >> 1; o > 0; o >>= 1) {
    if ((int)threadIdx.x < o) { sh[threadIdx.x] += sh[threadIdx.x + o];
                                sh2[threadIdx.x] += sh2[threadIdx.x + o]; }
    __syncthreads();
  }
  float m   = sh[0] / (float)C;
  float inv = rsqrtf(sh2[0] / (float)C - m * m + LN_EPS);
  for (int c = threadIdx.x; c < C; c += blockDim.x)
    y[t * (long)C + c] = (row[c] - m) * inv * g[c] + b[c];
}

// Depthwise causal conv (DCONV=4) along L + SiLU. x has row stride xld.
__global__ void k_dwconv_silu(const float* __restrict__ x, int xld,
                              const float* __restrict__ w, const float* __restrict__ b,
                              float* __restrict__ y, long S, int L, int C) {
  long i = (long)blockIdx.x * blockDim.x + threadIdx.x;
  if (i >= S * (long)L * C) return;
  int  c = (int)(i % C);
  long t = i / C;
  int  l = (int)(t % L);
  long s = t / L;
  float sum = b[c];
#pragma unroll
  for (int j = 0; j < 4; ++j) {
    int ll = l - 3 + j;
    if (ll >= 0) sum += x[(s * L + ll) * (long)xld + c] * w[c * 4 + j];
  }
  float sg = 1.f / (1.f + __expf(-sum));
  y[i] = sum * sg;
}

// Selective scan: per (seq, channel) thread, 16 states in registers.
// y = (sum_n h_n * C_n) + D*x, gated by silu(z).
__global__ void k_scan(const float* __restrict__ dt,
                       const float* __restrict__ xdbl, int xld, int dt_rank,
                       const float* __restrict__ xi,
                       const float* __restrict__ z, int zld,
                       const float* __restrict__ A_log, const float* __restrict__ Dp,
                       float* __restrict__ y, long S, int L, int C) {
  long idx = (long)blockIdx.x * blockDim.x + threadIdx.x;
  if (idx >= S * (long)C) return;
  long s = idx / C;
  int  d = (int)(idx % C);
  float a[16], h[16];
#pragma unroll
  for (int n = 0; n < 16; ++n) { a[n] = -__expf(A_log[d * 16 + n]); h[n] = 0.f; }
  float Dd = Dp[d];
  for (int l = 0; l < L; ++l) {
    long t   = s * L + l;
    float dv = dt[t * (long)C + d];
    float xv = xi[t * (long)C + d];
    const float* Bm = xdbl + t * (long)xld + dt_rank;
    const float* Cm = Bm + 16;
    float acc = 0.f;
#pragma unroll
    for (int n = 0; n < 16; ++n) {
      h[n] = h[n] * __expf(dv * a[n]) + dv * Bm[n] * xv;
      acc += h[n] * Cm[n];
    }
    float yv = acc + Dd * xv;
    float zv = z[t * (long)zld + d];
    yv *= zv / (1.f + __expf(-zv));
    y[t * (long)C + d] = yv;
  }
}

// =====================================================================
// Host-side orchestration
// =====================================================================
namespace {

struct Ctx {
  hipStream_t stream;
  char*  ws;
  size_t off;
  size_t cap;
  void* const* in;
};

inline void* ar_alloc(Ctx& c, size_t bytes) {
  size_t a = (c.off + 255) & ~(size_t)255;
  c.off = a + bytes;
  return c.ws + a;
}
inline float*     arF(Ctx& c, size_t elems) { return (float*)ar_alloc(c, elems * 4); }
inline _Float16*  arH(Ctx& c, size_t elems) { return (_Float16*)ar_alloc(c, elems * 2); }
inline const float* F32(Ctx& c, int i) { return (const float*)c.in[i]; }
inline unsigned nb(long total) { return (unsigned)((total + 255) / 256); }

void gemm_core(Ctx& cx, const _Float16* Ap, const _Float16* Bp,
               long nTi, int nTj2, int nKc,
               float* C, long M, int N, int ldc, const float* bias, int act) {
  long tiles = nTi * (long)nTj2;
  unsigned blocks = (unsigned)((tiles + 7) / 8);   // 256 thr = 8 waves/block
  k_gemm_wmma<<<blocks, 256, 0, cx.stream>>>(Ap, Bp, C, M, N, nKc, nTj2, ldc,
                                             bias, act, tiles);
}

// C[M,N] = A[M,K](f32,lda) * op(B), B given as [N,K] (transB) or [K,N].
void gemm(Ctx& cx, const float* A, long lda, const float* Bsrc, long ldb, bool transB,
          long M, int N, int K, float* C, int ldc, const float* bias, int act) {
  int  nKc  = ((K + 31) & ~31) / 32;
  long nTi  = ((M + 15) & ~15L) / 16;
  int  nTj  = ((N + 31) & ~31) / 16;   // padded to 32 cols -> even # of 16-tiles
  int  nTj2 = nTj / 2;
  size_t mk = cx.off;
  _Float16* Ap = arH(cx, (size_t)nTi * nKc * 512);
  _Float16* Bp = arH(cx, (size_t)nTj * nKc * 512);
  k_packfrag_a<<<nb(nTi * (long)nKc * 512), 256, 0, cx.stream>>>(A, lda, M, K, Ap, nKc, nTi);
  k_packfrag_b<<<nb((long)nTj * nKc * 512), 256, 0, cx.stream>>>(Bsrc, ldb, N, K,
                                                                 transB ? 1 : 0, Bp, nKc, nTj);
  gemm_core(cx, Ap, Bp, nTi, nTj2, nKc, C, M, N, ldc, bias, act);
  cx.off = mk;  // stream-ordered: safe to reuse
}

// 3x3 pad-1 conv on HWC via fragment-packed im2col + WMMA GEMM.
void conv3x3(Ctx& cx, const float* x, int H, int W, int Cin, const float* w,
             int Cout, float* out) {
  int  K    = Cin * 9;
  int  nKc  = ((K + 31) & ~31) / 32;
  long P    = (long)H * W;
  long nTi  = ((P + 15) & ~15L) / 16;
  int  nTj  = ((Cout + 31) & ~31) / 16;
  int  nTj2 = nTj / 2;
  size_t mk = cx.off;
  _Float16* Ap = arH(cx, (size_t)nTi * nKc * 512);
  _Float16* Bp = arH(cx, (size_t)nTj * nKc * 512);
  k_im2col_frag<<<nb(nTi * (long)nKc * 512), 256, 0, cx.stream>>>(x, H, W, Cin, Ap, nKc, nTi);
  k_packfrag_b<<<nb((long)nTj * nKc * 512), 256, 0, cx.stream>>>(w, K, Cout, K, 1, Bp, nKc, nTj);
  gemm_core(cx, Ap, Bp, nTi, nTj2, nKc, out, P, Cout, Cout, nullptr, 0);
  cx.off = mk;
}

void bn(Ctx& cx, float* x, long P, int C, const float* g, const float* b, int relu) {
  size_t mk = cx.off;
  float* mv = arF(cx, (size_t)2 * C);
  k_bn_stats<<<C, 256, 0, cx.stream>>>(x, P, C, mv, mv + C);
  k_bn_apply<<<nb(P * C), 256, 0, cx.stream>>>(x, P, C, mv, mv + C, g, b, relu);
  cx.off = mk;
}

// Mamba param leaf order (sorted keys): A_log, D, conv_b, conv_w, dt_b, dt_w,
// in_proj, out_proj, x_proj  -> offsets +0..+8 from base pb.
void run_mamba(Ctx& cx, int pb, const float* x, float* out, long S, int L, int D) {
  int d_inner = 2 * D;
  int R       = (D + 15) / 16;   // dt_rank = ceil(D/16)
  long T      = S * (long)L;
  size_t mk = cx.off;
  float* xz = arF(cx, (size_t)T * 2 * d_inner);
  gemm(cx, x, D, F32(cx, pb + 6), D, true, T, 2 * d_inner, D, xz, 2 * d_inner, nullptr, 0);
  float* xc = arF(cx, (size_t)T * d_inner);
  k_dwconv_silu<<<nb(T * (long)d_inner), 256, 0, cx.stream>>>(
      xz, 2 * d_inner, F32(cx, pb + 3), F32(cx, pb + 2), xc, S, L, d_inner);
  float* xdbl = arF(cx, (size_t)T * (R + 32));
  gemm(cx, xc, d_inner, F32(cx, pb + 8), d_inner, true, T, R + 32, d_inner,
       xdbl, R + 32, nullptr, 0);
  float* dt = arF(cx, (size_t)T * d_inner);
  gemm(cx, xdbl, R + 32, F32(cx, pb + 5), R, true, T, d_inner, R,
       dt, d_inner, F32(cx, pb + 4), /*softplus*/ 2);
  float* y = arF(cx, (size_t)T * d_inner);
  k_scan<<<nb(S * (long)d_inner), 256, 0, cx.stream>>>(
      dt, xdbl, R + 32, R, xc, xz + d_inner, 2 * d_inner,
      F32(cx, pb + 0), F32(cx, pb + 1), y, S, L, d_inner);
  gemm(cx, y, d_inner, F32(cx, pb + 7), d_inner, true, T, D, d_inner, out, D, nullptr, 0);
  cx.off = mk;
}

// m2d leaves (from mcb base mb): mh=mb+6..14, mv=mb+15..23,
// nh_b=mb+24, nh_w=mb+25, nv_b=mb+26, nv_w=mb+27
void run_mamba2d(Ctx& cx, int mb, const float* x, int H, int W, int C, float* out) {
  long P = (long)H * W;
  size_t mk = cx.off;
  float* xh = arF(cx, (size_t)P * C);
  {
    size_t m2 = cx.off;
    float* ln = arF(cx, (size_t)P * C);
    k_layernorm<<<(unsigned)P, 256, 0, cx.stream>>>(x, F32(cx, mb + 25), F32(cx, mb + 24), ln, C);
    run_mamba(cx, mb + 6, ln, xh, H, W, C);   // S=H, L=W
    cx.off = m2;
  }
  float* xv = arF(cx, (size_t)P * C);
  {
    size_t m2 = cx.off;
    float* xt = arF(cx, (size_t)P * C);
    float* ln = arF(cx, (size_t)P * C);
    k_transpose_hwc<<<nb(P * C), 256, 0, cx.stream>>>(x, xt, H, W, C);
    k_layernorm<<<(unsigned)P, 256, 0, cx.stream>>>(xt, F32(cx, mb + 27), F32(cx, mb + 26), ln, C);
    run_mamba(cx, mb + 15, ln, xv, W, H, C);  // S=W, L=H
    cx.off = m2;
  }
  float* xvb = arF(cx, (size_t)P * C);
  k_transpose_hwc<<<nb(P * C), 256, 0, cx.stream>>>(xv, xvb, W, H, C);
  k_add3<<<nb(P * C), 256, 0, cx.stream>>>(xh, xvb, x, out, P * C);
  cx.off = mk;
}

// dc leaves (base db): b1, b2, g1, g2, w1, w2 -> +0..+5
void run_double_conv(Ctx& cx, int db, const float* x, int H, int W,
                     int Cin, int Cout, float* out) {
  long P = (long)H * W;
  size_t mk = cx.off;
  float* t1 = arF(cx, (size_t)P * Cout);
  conv3x3(cx, x, H, W, Cin, F32(cx, db + 4), Cout, t1);
  bn(cx, t1, P, Cout, F32(cx, db + 2), F32(cx, db + 0), 1);
  conv3x3(cx, t1, H, W, Cout, F32(cx, db + 5), Cout, out);
  bn(cx, out, P, Cout, F32(cx, db + 3), F32(cx, db + 1), 1);
  cx.off = mk;
}

// mcb leaves (base mb): dc=+0..5, m2d=+6..27, pw1_b=+28, pw1_w=+29, pw2_b=+30,
// pw2_b2=+31, pw2_g=+32, pw2_w=+33, sc_b=+34, sc_g=+35, sc_w=+36
void run_mcb(Ctx& cx, int mb, const float* x, int H, int W,
             int Cin, int Cout, int mdim, float* out) {
  long P = (long)H * W;
  size_t mk = cx.off;
  float* idn = arF(cx, (size_t)P * Cout);
  gemm(cx, x, Cin, F32(cx, mb + 36), Cin, true, P, Cout, Cin, idn, Cout, nullptr, 0);
  bn(cx, idn, P, Cout, F32(cx, mb + 35), F32(cx, mb + 34), 0);
  float* dc = arF(cx, (size_t)P * Cout);
  run_double_conv(cx, mb + 0, x, H, W, Cin, Cout, dc);
  float* t = arF(cx, (size_t)P * mdim);
  gemm(cx, dc, Cout, F32(cx, mb + 29), Cout, true, P, mdim, Cout,
       t, mdim, F32(cx, mb + 28), /*relu*/ 1);
  float* m2 = arF(cx, (size_t)P * mdim);
  run_mamba2d(cx, mb, t, H, W, mdim, m2);
  float* t2 = arF(cx, (size_t)P * Cout);
  gemm(cx, m2, mdim, F32(cx, mb + 33), mdim, true, P, Cout, mdim,
       t2, Cout, F32(cx, mb + 30), 0);
  bn(cx, t2, P, Cout, F32(cx, mb + 32), F32(cx, mb + 31), 1);
  k_add2<<<nb(P * Cout), 256, 0, cx.stream>>>(t2, idn, out, P * Cout);
  cx.off = mk;
}

// up leaves (base ub): conv=+0..36, up_b=+37, up_w=+38
void run_up(Ctx& cx, int ub, const float* x1, int H1, int W1, int Cin,
            const float* x2, int Cout, int mdim, float* out) {
  int  Cmid = Cin / 2;
  long P1 = (long)H1 * W1;
  int  H2 = 2 * H1, W2 = 2 * W1;
  long P2 = (long)H2 * W2;
  size_t mk = cx.off;
  float* T = arF(cx, (size_t)P1 * Cmid * 4);
  // up_w is (Cin, Cmid, 2, 2): flatten to row-major [Cin x (Cmid*4)], no transpose.
  gemm(cx, x1, Cin, F32(cx, ub + 38), Cmid * 4, false, P1, Cmid * 4, Cin,
       T, Cmid * 4, nullptr, 0);
  float* up = arF(cx, (size_t)P2 * Cmid);
  k_up_scatter<<<nb(P1 * Cmid * 4), 256, 0, cx.stream>>>(T, F32(cx, ub + 37), up, H1, W1, Cmid);
  float* cat = arF(cx, (size_t)P2 * Cin);
  k_concat<<<nb(P2 * Cin), 256, 0, cx.stream>>>(x2, up, cat, P2, Cmid, Cmid);
  run_mcb(cx, ub + 0, cat, H2, W2, Cin, Cout, mdim, out);
  cx.off = mk;
}

} // namespace

// =====================================================================
// Entry point.
// Assumed flattened input order: setup_inputs() top-level insertion order
// (x first), then params flattened with JAX sorted-key order:
//   d_in[0]   = x (1,3,128,128)
//   d_in[1..] = down1(37), down2(37), down3(37), down4(37),
//               inc(6: b1,b2,g1,g2,w1,w2), outc_b, outc_w,
//               up1(39), up2(39), up3(39), up4(39)   -> 313 tensors total
// =====================================================================
extern "C" void kernel_launch(void* const* d_in, const int* in_sizes, int n_in,
                              void* d_out, int out_size, void* d_ws, size_t ws_size,
                              hipStream_t stream) {
  (void)in_sizes; (void)n_in; (void)out_size;
  Ctx cx{stream, (char*)d_ws, 0, ws_size, d_in};

  const int PX    = 1;            // params base
  const int down1 = PX + 0,   down2 = PX + 37,  down3 = PX + 74,  down4 = PX + 111;
  const int inc   = PX + 148;
  const int outcB = PX + 154, outcW = PX + 155;
  const int up1   = PX + 156, up2   = PX + 195, up3 = PX + 234,   up4   = PX + 273;
  const float* xin = F32(cx, 0);

  // ---- persistent buffers (skip connections + stage outputs) ----
  float* xh0 = arF(cx, 16384u * 3);
  float* x1  = arF(cx, 16384u * 32);
  float* x2b = arF(cx, 4096u * 64);
  float* x3  = arF(cx, 1024u * 128);
  float* x4  = arF(cx, 256u * 256);
  float* x5  = arF(cx, 64u * 512);
  float* p1  = arF(cx, 4096u * 32);
  float* p2  = arF(cx, 1024u * 64);
  float* p3  = arF(cx, 256u * 128);
  float* p4  = arF(cx, 64u * 256);
  float* y1  = arF(cx, 256u * 256);
  float* y2  = arF(cx, 1024u * 128);
  float* y3  = arF(cx, 4096u * 64);
  float* y4  = arF(cx, 16384u * 32);
  float* yo  = arF(cx, 16384u * 2);

  // ---- encoder ----
  k_nchw_to_hwc<<<nb(3L * 16384), 256, 0, stream>>>(xin, xh0, 3, 16384);
  run_double_conv(cx, inc, xh0, 128, 128, 3, 32, x1);

  k_maxpool2<<<nb(4096L * 32), 256, 0, stream>>>(x1, p1, 128, 128, 32);
  run_mcb(cx, down1, p1, 64, 64, 32, 64, 128, x2b);

  k_maxpool2<<<nb(1024L * 64), 256, 0, stream>>>(x2b, p2, 64, 64, 64);
  run_mcb(cx, down2, p2, 32, 32, 64, 128, 256, x3);

  k_maxpool2<<<nb(256L * 128), 256, 0, stream>>>(x3, p3, 32, 32, 128);
  run_mcb(cx, down3, p3, 16, 16, 128, 256, 512, x4);

  k_maxpool2<<<nb(64L * 256), 256, 0, stream>>>(x4, p4, 16, 16, 256);
  run_mcb(cx, down4, p4, 8, 8, 256, 512, 1024, x5);

  // ---- decoder ----
  run_up(cx, up1, x5, 8,  8,  512, x4,  256, 512, y1);
  run_up(cx, up2, y1, 16, 16, 256, x3,  128, 256, y2);
  run_up(cx, up3, y2, 32, 32, 128, x2b, 64,  128, y3);
  run_up(cx, up4, y3, 64, 64, 64,  x1,  32,  32,  y4);

  // ---- output head (1x1 conv) ----
  gemm(cx, y4, 32, F32(cx, outcW), 32, true, 16384, 2, 32, yo, 2, F32(cx, outcB), 0);
  k_hwc_to_nchw<<<nb(2L * 16384), 256, 0, stream>>>(yo, (float*)d_out, 2, 16384);
}
